// MultiHeadAttention_81166291960266
// MI455X (gfx1250) — matrix operand order repaired and checked by reading, not compiled, not run on hardware
//
#include <hip/hip_runtime.h>

#define BATCH   4
#define SEQ     2048
#define DMODEL  1024
#define NHEAD   16
#define DK      64
#define MROWS   (BATCH * SEQ)      // 8192

typedef __bf16 bf16_t;
typedef bf16_t v16bf __attribute__((ext_vector_type(16)));
typedef float  v8f   __attribute__((ext_vector_type(8)));

// int4 pointers qualified with the address spaces the async builtins expect:
// param 1 = global (addrspace(1)) int4*, param 2 = LDS (addrspace(3)) int4*.
typedef int v4i_t __attribute__((ext_vector_type(4)));
typedef __attribute__((address_space(1))) v4i_t* gptr_v4i;
typedef __attribute__((address_space(3))) v4i_t* lptr_v4i;

#if defined(__has_builtin)
#if __has_builtin(__builtin_amdgcn_global_load_async_to_lds_b128)
#define HAVE_ASYNC_LDS 1
#endif
#endif
#ifndef HAVE_ASYNC_LDS
#define HAVE_ASYNC_LDS 0
#endif

__device__ __forceinline__ void wait_asynccnt0() {
#if defined(__has_builtin) && __has_builtin(__builtin_amdgcn_s_wait_asynccnt)
    __builtin_amdgcn_s_wait_asynccnt(0);
#else
    asm volatile("s_wait_asynccnt 0x0" ::: "memory");
#endif
}

union BF16Frag {
    v16bf v;
    unsigned short h[16];
    uint4 q[2];
};

// fp32 -> bf16 round-to-nearest-even (in-register conversion; the GEMMs are
// compute-bound so we avoid any HBM-side conversion prepass).
__device__ __forceinline__ unsigned short f2bf(float f) {
    union { float f; unsigned u; } un; un.f = f;
    unsigned r = un.u + 0x7FFFu + ((un.u >> 16) & 1u);
    return (unsigned short)(r >> 16);
}

__device__ __forceinline__ void cvt8(const float4 a, const float4 b, unsigned short* h) {
    h[0] = f2bf(a.x); h[1] = f2bf(a.y); h[2] = f2bf(a.z); h[3] = f2bf(a.w);
    h[4] = f2bf(b.x); h[5] = f2bf(b.y); h[6] = f2bf(b.z); h[7] = f2bf(b.w);
}

// ---------------------------------------------------------------------------
// Y[m,n] = sum_k A[m,k] * W[n,k]   (X @ W^T), M=8192, N=1024, K=1024
// A: fp32 (projections) or bf16 (context for the O projection)
// Out: bf16 head-split (B,H,S,DK) for projections, or fp32 row-major (final).
// Block = 256 threads = 8 wave32 waves, block tile 128x64, wave tile 32x32.
// ---------------------------------------------------------------------------
template <bool A_BF16, bool OUT_F32>
__global__ __launch_bounds__(256) void wmma_gemm_xwT(
    const void* __restrict__ Av,
    const float* __restrict__ W,
    void* __restrict__ Outv,
    int M, int N, int K)
{
    const int lane = threadIdx.x & 31;
    const int lr   = lane & 15;       // row/col within 16-tile
    const int hi   = lane >> 4;       // half-wave select
    const int wid  = threadIdx.x >> 5;
    const int rowBase = blockIdx.x * 128 + (wid & 3) * 32;
    const int colBase = blockIdx.y * 64  + (wid >> 2) * 32;

    const float*          Af = (const float*)Av;
    const unsigned short* Ab = (const unsigned short*)Av;

    v8f acc[2][2];
    const v8f vzero = {0.f, 0.f, 0.f, 0.f, 0.f, 0.f, 0.f, 0.f};
#pragma unroll
    for (int mi = 0; mi < 2; ++mi)
#pragma unroll
        for (int ni = 0; ni < 2; ++ni) acc[mi][ni] = vzero;

    for (int k0 = 0; k0 < K; k0 += 32) {
        BF16Frag a[2], bw[2];
#pragma unroll
        for (int mi = 0; mi < 2; ++mi) {
            const int row = rowBase + mi * 16 + lr;
            if (A_BF16) {
                const unsigned short* Ar = Ab + (size_t)row * K + k0;
                a[mi].q[0] = *(const uint4*)(Ar + hi * 8);
                a[mi].q[1] = *(const uint4*)(Ar + 16 + hi * 8);
            } else {
                const float* Ar = Af + (size_t)row * K + k0;
                cvt8(*(const float4*)(Ar + hi * 8),
                     *(const float4*)(Ar + hi * 8 + 4),  &a[mi].h[0]);
                cvt8(*(const float4*)(Ar + 16 + hi * 8),
                     *(const float4*)(Ar + 16 + hi * 8 + 4), &a[mi].h[8]);
            }
        }
#pragma unroll
        for (int ni = 0; ni < 2; ++ni) {
            const int n = colBase + ni * 16 + lr;
            const float* Wr = W + (size_t)n * K + k0 + hi * 8;
            cvt8(*(const float4*)(Wr),     *(const float4*)(Wr + 4),  &bw[ni].h[0]);
            cvt8(*(const float4*)(Wr + 16), *(const float4*)(Wr + 20), &bw[ni].h[8]);
        }
#pragma unroll
        for (int mi = 0; mi < 2; ++mi)
#pragma unroll
            for (int ni = 0; ni < 2; ++ni)
                acc[mi][ni] = __builtin_amdgcn_wmma_f32_16x16x32_bf16(
                    false, a[mi].v, false, bw[ni].v, (short)0, acc[mi][ni],
                    false, false);
    }

#pragma unroll
    for (int mi = 0; mi < 2; ++mi)
#pragma unroll
        for (int ni = 0; ni < 2; ++ni)
#pragma unroll
            for (int r = 0; r < 8; ++r) {
                const int m = rowBase + mi * 16 + r + hi * 8;
                const int n = colBase + ni * 16 + lr;
                if (OUT_F32) {
                    ((float*)Outv)[(size_t)m * N + n] = acc[mi][ni][r];
                } else {
                    // head-split (B,H,S,DK): m = b*SEQ+s ; n = h*DK+d
                    const int bb = m >> 11, s = m & (SEQ - 1);
                    const int hh = n >> 6,  d = n & (DK - 1);
                    ((unsigned short*)Outv)
                        [(((size_t)(bb * NHEAD + hh) * SEQ + s) << 6) + d] =
                        f2bf(acc[mi][ni][r]);
                }
            }
}

// ---------------------------------------------------------------------------
// Flash attention: one block = one (b,h) x 128 q-rows. 8 waves x 16 rows.
// K tile staged Global->LDS via the CDNA5 async pipe (ASYNCcnt) -- no VGPR
// round-trip; V staged transposed through registers (ds_store).
// Online softmax; P re-shaped C-layout -> A-layout via a per-wave LDS slab.
// ---------------------------------------------------------------------------
__global__ __launch_bounds__(256) void flash_attn(
    const unsigned short* __restrict__ Qh,
    const unsigned short* __restrict__ Kh,
    const unsigned short* __restrict__ Vh,
    const int* __restrict__ mask,
    unsigned short* __restrict__ Ctx)
{
    __shared__ __align__(16) unsigned short Ks[64 * 64];      // [key][d]
    __shared__ __align__(16) unsigned short Vs[64 * 64];      // [d][key]
    __shared__ __align__(16) unsigned short Ps[8 * 16 * 64];  // per-wave P

    const int bh   = blockIdx.x;            // 0..63
    const int b    = bh >> 4;
    const int h    = bh & (NHEAD - 1);
    const int wid  = threadIdx.x >> 5;
    const int lane = threadIdx.x & 31;
    const int lr   = lane & 15;
    const int hi   = lane >> 4;
    const int qBase = blockIdx.y * 128 + wid * 16;

    const unsigned short* Qp = Qh + (size_t)bh * SEQ * DK;
    const unsigned short* Kp = Kh + (size_t)bh * SEQ * DK;
    const unsigned short* Vp = Vh + (size_t)bh * SEQ * DK;
    const int* mp = mask + b * SEQ;

    // Q fragments (A-layout), held in registers for the whole key loop.
    BF16Frag qa[2];
    {
        const unsigned short* Qrow = Qp + (size_t)(qBase + lr) * DK;
#pragma unroll
        for (int s = 0; s < 2; ++s) {
            qa[s].q[0] = *(const uint4*)(Qrow + s * 32 + hi * 8);
            qa[s].q[1] = *(const uint4*)(Qrow + s * 32 + 16 + hi * 8);
        }
    }

    const v8f vzero = {0.f, 0.f, 0.f, 0.f, 0.f, 0.f, 0.f, 0.f};
    v8f o[4];
#pragma unroll
    for (int ni = 0; ni < 4; ++ni) o[ni] = vzero;
    float mrow[8], lrow[8];
#pragma unroll
    for (int r = 0; r < 8; ++r) { mrow[r] = -3.0e38f; lrow[r] = 0.f; }

    unsigned short* Pw = &Ps[wid * 16 * 64];

    const int t    = threadIdx.x;
    const int krow = t >> 2;            // 0..63
    const int dseg = (t & 3) << 4;      // 0,16,32,48

    for (int kt = 0; kt < SEQ / 64; ++kt) {
        const int kbase = kt * 64;

        // ---- stage K tile: async Global->LDS (ASYNCcnt, no VGPR round-trip)
        {
#if HAVE_ASYNC_LDS
            unsigned short* gk0 =
                const_cast<unsigned short*>(Kp + (size_t)(kbase + krow) * DK + dseg);
            __builtin_amdgcn_global_load_async_to_lds_b128(
                (gptr_v4i)gk0,       (lptr_v4i)&Ks[krow * 64 + dseg],     0, 0);
            __builtin_amdgcn_global_load_async_to_lds_b128(
                (gptr_v4i)(gk0 + 8), (lptr_v4i)&Ks[krow * 64 + dseg + 8], 0, 0);
#else
            const uint4* gk = (const uint4*)(Kp + (size_t)(kbase + krow) * DK + dseg);
            uint4 a0 = gk[0], a1 = gk[1];
            *(uint4*)(&Ks[krow * 64 + dseg])     = a0;
            *(uint4*)(&Ks[krow * 64 + dseg + 8]) = a1;
#endif
            // ---- stage V tile transposed ([d][key]) via registers ----
            const uint4* gv = (const uint4*)(Vp + (size_t)(kbase + krow) * DK + dseg);
            unsigned short vl[16];
            *(uint4*)(&vl[0]) = gv[0];
            *(uint4*)(&vl[8]) = gv[1];
#pragma unroll
            for (int j = 0; j < 16; ++j) Vs[(dseg + j) * 64 + krow] = vl[j];

            if (kt + 1 < SEQ / 64) {   // global_prefetch_b8 of next tile
                __builtin_prefetch(Kp + (size_t)(kbase + 64 + krow) * DK + dseg, 0, 1);
                __builtin_prefetch(Vp + (size_t)(kbase + 64 + krow) * DK + dseg, 0, 1);
            }
        }
#if HAVE_ASYNC_LDS
        wait_asynccnt0();
#endif
        __syncthreads();

        // S = Q * K^T  (16 q-rows x 64 keys per wave)
        v8f sacc[4];
#pragma unroll
        for (int ni = 0; ni < 4; ++ni) sacc[ni] = vzero;
#pragma unroll
        for (int s = 0; s < 2; ++s)
#pragma unroll
            for (int ni = 0; ni < 4; ++ni) {
                BF16Frag kb;
                const unsigned short* kr =
                    &Ks[(ni * 16 + lr) * 64 + s * 32 + hi * 8];
                kb.q[0] = *(const uint4*)kr;
                kb.q[1] = *(const uint4*)(kr + 16);
                sacc[ni] = __builtin_amdgcn_wmma_f32_16x16x32_bf16(
                    false, qa[s].v, false, kb.v, (short)0, sacc[ni],
                    false, false);
            }

        // scale 1/sqrt(dk) then mask
#pragma unroll
        for (int ni = 0; ni < 4; ++ni) {
#pragma unroll
            for (int r = 0; r < 8; ++r) sacc[ni][r] = sacc[ni][r] * 0.125f;
            const int mv = mp[kbase + ni * 16 + lr];
            if (mv == 0)
#pragma unroll
                for (int r = 0; r < 8; ++r) sacc[ni][r] = -1.0e30f;
        }

        // online softmax; a C-layout row lives across 16 lanes of a half-wave
#pragma unroll
        for (int r = 0; r < 8; ++r) {
            float mx = fmaxf(fmaxf(sacc[0][r], sacc[1][r]),
                             fmaxf(sacc[2][r], sacc[3][r]));
#pragma unroll
            for (int off = 1; off < 16; off <<= 1)
                mx = fmaxf(mx, __shfl_xor(mx, off, 32));
            const float nm    = fmaxf(mrow[r], mx);
            const float alpha = __expf(mrow[r] - nm);
            float ls = 0.f;
#pragma unroll
            for (int ni = 0; ni < 4; ++ni) {
                const float p = __expf(sacc[ni][r] - nm);
                sacc[ni][r] = p;
                ls += p;
            }
#pragma unroll
            for (int off = 1; off < 16; off <<= 1)
                ls += __shfl_xor(ls, off, 32);
            lrow[r] = lrow[r] * alpha + ls;
            mrow[r] = nm;
#pragma unroll
            for (int ni = 0; ni < 4; ++ni) o[ni][r] = o[ni][r] * alpha;
        }

        // P (C-layout) -> per-wave LDS slab as bf16 [qrow][key]
#pragma unroll
        for (int ni = 0; ni < 4; ++ni)
#pragma unroll
            for (int r = 0; r < 8; ++r)
                Pw[(r + hi * 8) * 64 + ni * 16 + lr] = f2bf(sacc[ni][r]);

        // O += P * V
#pragma unroll
        for (int s = 0; s < 2; ++s) {
            BF16Frag pa;
            const unsigned short* pr = Pw + lr * 64 + s * 32;
            pa.q[0] = *(const uint4*)(pr + hi * 8);
            pa.q[1] = *(const uint4*)(pr + 16 + hi * 8);
#pragma unroll
            for (int ni = 0; ni < 4; ++ni) {
                BF16Frag vb;
                const unsigned short* vr =
                    &Vs[(ni * 16 + lr) * 64 + s * 32 + hi * 8];
                vb.q[0] = *(const uint4*)vr;
                vb.q[1] = *(const uint4*)(vr + 16);
                o[ni] = __builtin_amdgcn_wmma_f32_16x16x32_bf16(
                    false, pa.v, false, vb.v, (short)0, o[ni], false, false);
            }
        }
        __syncthreads();
    }

    // normalize and write context as bf16 in (B,S,H*DK) for the O projection
#pragma unroll
    for (int r = 0; r < 8; ++r) {
        const float inv = 1.0f / lrow[r];
        const int qg = qBase + r + hi * 8;
        const size_t rowoff = ((size_t)(b * SEQ + qg)) * DMODEL + h * DK;
#pragma unroll
        for (int ni = 0; ni < 4; ++ni)
            Ctx[rowoff + ni * 16 + lr] = f2bf(o[ni][r] * inv);
    }
}

extern "C" void kernel_launch(void* const* d_in, const int* in_sizes, int n_in,
                              void* d_out, int out_size, void* d_ws, size_t ws_size,
                              hipStream_t stream) {
    (void)in_sizes; (void)n_in; (void)out_size; (void)ws_size;

    const float* q    = (const float*)d_in[0];
    const float* k    = (const float*)d_in[1];
    const float* v    = (const float*)d_in[2];
    const int*   mask = (const int*)d_in[3];
    const float* Wq   = (const float*)d_in[4];
    const float* Wk   = (const float*)d_in[5];
    const float* Wv   = (const float*)d_in[6];
    const float* Wo   = (const float*)d_in[7];
    float* out = (float*)d_out;

    // Workspace: bf16 Q,K,V in (B,H,S,DK) + bf16 context (B,S,DMODEL) = 64 MiB
    const size_t elems = (size_t)BATCH * NHEAD * SEQ * DK;  // 8,388,608
    unsigned short* Qh  = (unsigned short*)d_ws;
    unsigned short* Kh  = Qh + elems;
    unsigned short* Vh  = Kh + elems;
    unsigned short* Ctx = Vh + elems;

    dim3 gg(MROWS / 128, DMODEL / 64);   // 64 x 16
    dim3 blk(256);

    wmma_gemm_xwT<false, false><<<gg, blk, 0, stream>>>(q, Wq, Qh, MROWS, DMODEL, DMODEL);
    wmma_gemm_xwT<false, false><<<gg, blk, 0, stream>>>(k, Wk, Kh, MROWS, DMODEL, DMODEL);
    wmma_gemm_xwT<false, false><<<gg, blk, 0, stream>>>(v, Wv, Vh, MROWS, DMODEL, DMODEL);

    dim3 ga(BATCH * NHEAD, SEQ / 128);   // 64 x 16
    flash_attn<<<ga, blk, 0, stream>>>(Qh, Kh, Vh, mask, Ctx);

    wmma_gemm_xwT<true, true><<<gg, blk, 0, stream>>>(Ctx, Wo, out, MROWS, DMODEL, DMODEL);
}
